// MixtureOfExperts_81509889344107
// MI455X (gfx1250) — matrix operand order repaired and checked by reading, not compiled, not run on hardware
//
#include <hip/hip_runtime.h>
#include <hip/hip_bf16.h>
#include <math.h>

// MoE dims (fixed by the reference)
#define E_ 8
#define D_ 1024
#define F_ 4096
#define T_ 4096   // B*S = 4*1024
#define MT_ 32    // tokens per expert tile (2 M-halves of 16)

typedef __bf16 bf16_t;
typedef __attribute__((ext_vector_type(16))) __bf16 v16bf;
typedef __attribute__((ext_vector_type(8)))  float  v8f;

union FragBF { v16bf v; uint4 u[2]; };

// gelu_new(v) = 0.5*v*(1+tanh(0.79788456*(v+0.044715 v^3))) = v * sigmoid(2s)
// branch-free: one v_exp_f32 + one v_rcp_f32, saturates correctly at +/-inf
__device__ __forceinline__ float gelu_new(float v) {
  float s2 = 1.5957691216057308f * (v + 0.044715f * v * v * v);  // 2*s
  return v * __builtin_amdgcn_rcpf(1.f + __expf(-s2));
}

// ---------------- router: logits, softmax, top-2, compaction ----------------
__global__ void router_kernel(const float* __restrict__ x,
                              const float* __restrict__ gw,
                              float* __restrict__ logits,
                              int* __restrict__ counts,
                              int* __restrict__ tok_ids,
                              float* __restrict__ tok_w) {
  int t = blockIdx.x * blockDim.x + threadIdx.x;
  if (t >= T_) return;
  const float* xr = x + (size_t)t * D_;
  float acc[E_];
#pragma unroll
  for (int e = 0; e < E_; ++e) acc[e] = 0.f;
  for (int d = 0; d < D_; ++d) {
    float xv = xr[d];
#pragma unroll
    for (int e = 0; e < E_; ++e) acc[e] = fmaf(xv, gw[d * E_ + e], acc[e]);
  }
#pragma unroll
  for (int e = 0; e < E_; ++e) logits[(size_t)t * E_ + e] = acc[e];

  float mx = acc[0];
#pragma unroll
  for (int e = 1; e < E_; ++e) mx = fmaxf(mx, acc[e]);
  float p[E_];
#pragma unroll
  for (int e = 0; e < E_; ++e) p[e] = __expf(acc[e] - mx);

  // top-2, lowest index wins ties (matches lax.top_k)
  int i0 = 0;
#pragma unroll
  for (int e = 1; e < E_; ++e) if (p[e] > p[i0]) i0 = e;
  int i1 = (i0 == 0) ? 1 : 0;
#pragma unroll
  for (int e = 0; e < E_; ++e) if (e != i0 && p[e] > p[i1]) i1 = e;

  float rden = __builtin_amdgcn_rcpf(p[i0] + p[i1]);
  int s0 = atomicAdd(&counts[i0], 1);
  tok_ids[i0 * T_ + s0] = t; tok_w[i0 * T_ + s0] = p[i0] * rden;
  int s1 = atomicAdd(&counts[i1], 1);
  tok_ids[i1 * T_ + s1] = t; tok_w[i1 * T_ + s1] = p[i1] * rden;
}

// ---------------- fp32 -> bf16 conversion (K-contiguous layouts) ----------------
__global__ void cvt_x_kernel(const float* __restrict__ x, bf16_t* __restrict__ xb, int n) {
  int i = blockIdx.x * blockDim.x + threadIdx.x;
  if (i < n) xb[i] = (bf16_t)x[i];
}

// w1t[e][f][d] = fc_w[e][d][f]   (GEMM1 B-operand, K=d contiguous)
__global__ void cvt_w1_kernel(const float* __restrict__ fcw, bf16_t* __restrict__ w1t) {
  size_t i = (size_t)blockIdx.x * blockDim.x + threadIdx.x;
  if (i >= (size_t)E_ * F_ * D_) return;
  int d = (int)(i % D_);
  size_t r = i / D_;
  int f = (int)(r % F_);
  int e = (int)(r / F_);
  w1t[i] = (bf16_t)fcw[((size_t)e * D_ + d) * F_ + f];
}

// w2t[e][d][f] = proj_w[e][f][d] (GEMM2 B-operand, K=f contiguous)
__global__ void cvt_w2_kernel(const float* __restrict__ pw, bf16_t* __restrict__ w2t) {
  size_t i = (size_t)blockIdx.x * blockDim.x + threadIdx.x;
  if (i >= (size_t)E_ * D_ * F_) return;
  int f = (int)(i % F_);
  size_t r = i / F_;
  int d = (int)(r % D_);
  int e = (int)(r / D_);
  w2t[i] = (bf16_t)pw[((size_t)e * F_ + f) * D_ + d];
}

// ---------------- fused expert MLP: fc -> gelu -> proj -> scatter-add ----------------
// 32-token tile per block, 8 waves; h tile (32 x 4096 bf16 = 256 KB) lives in LDS.
__global__ __launch_bounds__(256) void moe_expert_kernel(
    const bf16_t* __restrict__ xb,
    const bf16_t* __restrict__ w1t,
    const bf16_t* __restrict__ w2t,
    const float* __restrict__ fc_b,
    const float* __restrict__ proj_b,
    const int*  __restrict__ counts,
    const int*  __restrict__ tok_ids,
    const float* __restrict__ tok_w,
    float* __restrict__ out) {
  __shared__ bf16_t sH[MT_ * F_];   // 256 KB of the 320 KB WGP LDS
  __shared__ int    sTok[MT_];

  const int e    = blockIdx.y;
  const int tile = blockIdx.x;
  const int cnt  = counts[e];
  if (tile * MT_ >= cnt) return;              // block-uniform exit

  const int tid   = threadIdx.x;
  const int lane  = tid & 31;
  const int wave  = tid >> 5;                 // 8 waves
  const int nlane = lane & 15;
  const int hi    = lane >> 4;

  if (tid < MT_) {
    int idx = tile * MT_ + tid;
    sTok[tid] = (idx < cnt) ? tok_ids[e * T_ + idx] : tok_ids[e * T_]; // pad -> valid row
  }
  __syncthreads();

  // per-lane A row pointers (token rows, bf16, K-contiguous): fragments are 2x b128
  const bf16_t* rowA0 = xb + (size_t)sTok[nlane] * D_;
  const bf16_t* rowA1 = xb + (size_t)sTok[nlane + 16] * D_;

  const bf16_t* W1 = w1t + (size_t)e * F_ * D_;  // [F][D]
  const float*  b1 = fc_b + e * F_;

  // Phase 1: h[32 x F] = gelu_new(A @ W1 + b1) -> bf16 into sH
#pragma unroll 1
  for (int nt = 0; nt < F_ / (8 * 16); ++nt) {          // 32 N-tiles per wave
    const int n0 = wave * (F_ / 8) + nt * 16;
    // prefetch next N-tile's weight rows (global_prefetch_b8)
    if (nt + 1 < F_ / (8 * 16))
      __builtin_prefetch(W1 + (size_t)(n0 + 16 + nlane) * D_, 0, 1);
    v8f acc0 = {}, acc1 = {};
#pragma unroll 2
    for (int kt = 0; kt < D_ / 32; ++kt) {              // 32 K-steps
      FragBF fa0, fa1, fb;
      const int ka = kt * 32 + hi * 8;
      fa0.u[0] = *(const uint4*)(rowA0 + ka);
      fa0.u[1] = *(const uint4*)(rowA0 + ka + 16);
      fa1.u[0] = *(const uint4*)(rowA1 + ka);
      fa1.u[1] = *(const uint4*)(rowA1 + ka + 16);
      const bf16_t* pb = W1 + (size_t)(n0 + nlane) * D_ + kt * 32 + hi * 8;
      fb.u[0] = *(const uint4*)(pb);
      fb.u[1] = *(const uint4*)(pb + 16);
      acc0 = __builtin_amdgcn_wmma_f32_16x16x32_bf16(false, fa0.v, false, fb.v,
                                                     (short)0, acc0, false, false);
      acc1 = __builtin_amdgcn_wmma_f32_16x16x32_bf16(false, fa1.v, false, fb.v,
                                                     (short)0, acc1, false, false);
    }
    float bias = b1[n0 + nlane];
#pragma unroll
    for (int r = 0; r < 8; ++r) {
      int m = r + hi * 8;
      sH[m * F_ + (n0 + nlane)]        = (bf16_t)gelu_new(acc0[r] + bias);
      sH[(m + 16) * F_ + (n0 + nlane)] = (bf16_t)gelu_new(acc1[r] + bias);
    }
  }
  __syncthreads();

  const bf16_t* W2 = w2t + (size_t)e * D_ * F_;  // [D][F]
  const float*  b2 = proj_b + e * D_;
  const bf16_t* rowH0 = sH + nlane * F_;
  const bf16_t* rowH1 = sH + (nlane + 16) * F_;

  // Phase 2: out_tile[32 x D] = h @ W2 + b2, scaled by routing weight, scatter-add
#pragma unroll 1
  for (int nt = 0; nt < D_ / (8 * 16); ++nt) {          // 8 N-tiles per wave
    const int n0 = wave * (D_ / 8) + nt * 16;
    if (nt + 1 < D_ / (8 * 16))
      __builtin_prefetch(W2 + (size_t)(n0 + 16 + nlane) * F_, 0, 1);
    v8f acc0 = {}, acc1 = {};
#pragma unroll 2
    for (int kt = 0; kt < F_ / 32; ++kt) {              // 128 K-steps
      FragBF fa0, fa1, fb;
      const int ka = kt * 32 + hi * 8;
      fa0.u[0] = *(const uint4*)(rowH0 + ka);           // ds_load_b128
      fa0.u[1] = *(const uint4*)(rowH0 + ka + 16);
      fa1.u[0] = *(const uint4*)(rowH1 + ka);
      fa1.u[1] = *(const uint4*)(rowH1 + ka + 16);
      const bf16_t* pb = W2 + (size_t)(n0 + nlane) * F_ + kt * 32 + hi * 8;
      fb.u[0] = *(const uint4*)(pb);                    // global_load_b128
      fb.u[1] = *(const uint4*)(pb + 16);
      acc0 = __builtin_amdgcn_wmma_f32_16x16x32_bf16(false, fa0.v, false, fb.v,
                                                     (short)0, acc0, false, false);
      acc1 = __builtin_amdgcn_wmma_f32_16x16x32_bf16(false, fa1.v, false, fb.v,
                                                     (short)0, acc1, false, false);
    }
    float bias = b2[n0 + nlane];
#pragma unroll
    for (int r = 0; r < 8; ++r) {
      int m = r + hi * 8;
      int g0 = tile * MT_ + m;
      if (g0 < cnt) {
        float w = tok_w[e * T_ + g0];
        int  tk = tok_ids[e * T_ + g0];
        atomicAdd(out + (size_t)tk * D_ + (n0 + nlane), (acc0[r] + bias) * w);
      }
      int g1 = g0 + 16;
      if (g1 < cnt) {
        float w = tok_w[e * T_ + g1];
        int  tk = tok_ids[e * T_ + g1];
        atomicAdd(out + (size_t)tk * D_ + (n0 + nlane), (acc1[r] + bias) * w);
      }
    }
  }
}

// ---------------- launch ----------------
extern "C" void kernel_launch(void* const* d_in, const int* in_sizes, int n_in,
                              void* d_out, int out_size, void* d_ws, size_t ws_size,
                              hipStream_t stream) {
  (void)in_sizes; (void)n_in; (void)out_size; (void)ws_size;
  const float* x   = (const float*)d_in[0];  // [T, D]
  const float* gw  = (const float*)d_in[1];  // [D, E]
  const float* fcw = (const float*)d_in[2];  // [E, D, F]
  const float* fcb = (const float*)d_in[3];  // [E, F]
  const float* pw  = (const float*)d_in[4];  // [E, F, D]
  const float* pb  = (const float*)d_in[5];  // [E, D]

  float* out    = (float*)d_out;                 // [T*D] out, then [T*E] logits
  float* logits = out + (size_t)T_ * D_;

  char* ws = (char*)d_ws;
  bf16_t* xb  = (bf16_t*)ws;  ws += (size_t)T_ * D_ * sizeof(bf16_t);
  bf16_t* w1t = (bf16_t*)ws;  ws += (size_t)E_ * F_ * D_ * sizeof(bf16_t);
  bf16_t* w2t = (bf16_t*)ws;  ws += (size_t)E_ * D_ * F_ * sizeof(bf16_t);
  int*   counts = (int*)ws;   ws += 256;
  int*   tok_ids = (int*)ws;  ws += (size_t)E_ * T_ * sizeof(int);
  float* tok_w = (float*)ws;  ws += (size_t)E_ * T_ * sizeof(float);

  hipMemsetAsync(out, 0, (size_t)T_ * D_ * sizeof(float), stream);
  hipMemsetAsync(counts, 0, E_ * sizeof(int), stream);

  { int n = T_ * D_;
    cvt_x_kernel<<<(n + 255) / 256, 256, 0, stream>>>(x, xb, n); }
  { size_t n = (size_t)E_ * F_ * D_;
    cvt_w1_kernel<<<(unsigned)((n + 255) / 256), 256, 0, stream>>>(fcw, w1t); }
  { size_t n = (size_t)E_ * D_ * F_;
    cvt_w2_kernel<<<(unsigned)((n + 255) / 256), 256, 0, stream>>>(pw, w2t); }

  router_kernel<<<T_ / 256, 256, 0, stream>>>(x, gw, logits, counts, tok_ids, tok_w);

  dim3 grid(T_ / MT_, E_);   // worst case: all T tokens on one expert
  moe_expert_kernel<<<grid, 256, 0, stream>>>(xb, w1t, w2t, fcb, pb,
                                              counts, tok_ids, tok_w, out);
}
